// proxi_sampler_69526930588007
// MI455X (gfx1250) — compile-verified, matrix operand order checked by reading, not run on hardware
//
#include <hip/hip_runtime.h>
#include <hip/hip_bf16.h>
#include <stdint.h>

#define BATCH 1024
#define NOBJ  128
#define RREL  32
#define DDIM  512
#define CCLS  7
#define MROWS (BATCH * RREL) /* 32768 */

typedef __attribute__((ext_vector_type(16))) __bf16 v16bf;
typedef __attribute__((ext_vector_type(8)))  float  v8f;

union Frag16 { v16bf bf; unsigned short us[16]; uint4 q[2]; };

__device__ __forceinline__ unsigned short f2bf(float f) {
  union { float f; uint32_t u; } c; c.f = f;
  uint32_t u = c.u;
  return (unsigned short)((u + 0x7FFFu + ((u >> 16) & 1u)) >> 16);
}
__device__ __forceinline__ float bf2f(unsigned short h) {
  union { uint32_t u; float f; } c; c.u = ((uint32_t)h) << 16;
  return c.f;
}

// ---------------------------------------------------------------------------
// Prep: f32 (K x N) weight -> bf16 transposed (N x K) for contiguous B-frag runs
// ---------------------------------------------------------------------------
__global__ void transpose_bf16_kernel(const float* __restrict__ src,
                                      unsigned short* __restrict__ dst,
                                      int K, int N) {
  int idx = blockIdx.x * blockDim.x + threadIdx.x;
  if (idx < K * N) {
    int k = idx / N, n = idx % N;
    dst[(size_t)n * K + k] = f2bf(src[idx]);
  }
}

// ---------------------------------------------------------------------------
// Stage 1: init_rel = concat(obj[p0], obj[p1]) @ fuse_w + fuse_b
// M=32768, K=1024, N=512. Block = 16 rows x full N. A tile gathered + converted
// cooperatively into LDS once, then consumed by 8 waves via ds_load_b128.
// ---------------------------------------------------------------------------
__global__ __launch_bounds__(256)
void fuse_gemm_kernel(const float* __restrict__ obj, const int* __restrict__ pairs,
                      const unsigned short* __restrict__ WT,  // (512 x 1024) bf16
                      const float* __restrict__ bias,
                      unsigned short* __restrict__ initrel) { // (32768 x 512) bf16
  constexpr int KD = 2 * DDIM;        // 1024
  constexpr int KP = KD + 8;          // padded row (bank spread, 16B-aligned rows)
  __shared__ unsigned short sA[16 * KP];

  const int tid = threadIdx.x;
  const int mt  = blockIdx.x;

  // ---- cooperative gather + f32->bf16 convert into LDS (8-elem chunks) ----
  for (int c = tid; c < 16 * (KD / 8); c += 256) {
    const int row = c / (KD / 8);
    const int k   = (c % (KD / 8)) * 8;
    const int m   = mt * 16 + row;
    const int b   = m >> 5;
    const int r   = m & 31;
    const int p0  = pairs[(b * RREL + r) * 2 + 0];
    const int p1  = pairs[(b * RREL + r) * 2 + 1];
    const float* s = (k < DDIM)
        ? obj + ((size_t)b * NOBJ + p0) * DDIM + k
        : obj + ((size_t)b * NOBJ + p1) * DDIM + (k - DDIM);
    float4 x = ((const float4*)s)[0], y = ((const float4*)s)[1];
    unsigned short* d = &sA[row * KP + k];
    d[0]=f2bf(x.x); d[1]=f2bf(x.y); d[2]=f2bf(x.z); d[3]=f2bf(x.w);
    d[4]=f2bf(y.x); d[5]=f2bf(y.y); d[6]=f2bf(y.z); d[7]=f2bf(y.w);
  }
  __syncthreads();

  const int lane = tid & 31;
  const int l16  = lane & 15;
  const int h    = lane >> 4;
  const int nt   = tid >> 5;          // 8 waves -> 8 n-tiles of 64

  const v8f z = {0.f,0.f,0.f,0.f,0.f,0.f,0.f,0.f};
  v8f accs[4] = {z, z, z, z};

  for (int k = 0; k < KD; k += 32) {
    const int ka = k + 8 * h;
    const int kb = k + 16 + 8 * h;
    Frag16 a;
    a.q[0] = *(const uint4*)(&sA[l16 * KP + ka]);
    a.q[1] = *(const uint4*)(&sA[l16 * KP + kb]);
#pragma unroll
    for (int j = 0; j < 4; ++j) {
      Frag16 bf;
      const unsigned short* w = WT + (size_t)(nt * 64 + j * 16 + l16) * KD;
      bf.q[0] = *(const uint4*)(w + ka);
      bf.q[1] = *(const uint4*)(w + kb);
      accs[j] = __builtin_amdgcn_wmma_f32_16x16x32_bf16(false, a.bf, false, bf.bf,
                                                        (short)0, accs[j], false, false);
    }
  }
#pragma unroll
  for (int j = 0; j < 4; ++j) {
    const int col = nt * 64 + j * 16 + l16;
    const float bv = bias[col];
#pragma unroll
    for (int i = 0; i < 8; ++i) {
      const int row = mt * 16 + 8 * h + i;
      initrel[(size_t)row * DDIM + col] = f2bf(accs[j][i] + bv);
    }
  }
}

// ---------------------------------------------------------------------------
// Stage 2: rel_feat = relu((obj[p0]+obj[p1]) @ gcn_w + gcn_b) + init_rel
// Only relation rows of H are used, so A@X collapses to a 2-row gather-sum.
// M=32768, K=512, N=512. Same LDS-staged block structure.
// ---------------------------------------------------------------------------
__global__ __launch_bounds__(256)
void gcn_gemm_kernel(const float* __restrict__ obj, const int* __restrict__ pairs,
                     const unsigned short* __restrict__ WT,   // (512 x 512) bf16
                     const float* __restrict__ bias,
                     const unsigned short* __restrict__ initrel,
                     unsigned short* __restrict__ relfeat) {  // (32768 x 512) bf16
  constexpr int KD = DDIM;            // 512
  constexpr int KP = KD + 8;
  __shared__ unsigned short sA[16 * KP];

  const int tid = threadIdx.x;
  const int mt  = blockIdx.x;

  for (int c = tid; c < 16 * (KD / 8); c += 256) {
    const int row = c / (KD / 8);
    const int k   = (c % (KD / 8)) * 8;
    const int m   = mt * 16 + row;
    const int b   = m >> 5;
    const int r   = m & 31;
    const int p0  = pairs[(b * RREL + r) * 2 + 0];
    const int p1  = pairs[(b * RREL + r) * 2 + 1];
    const float sc = (p0 == p1) ? 0.f : 1.f;   // A entries collapse when p0==p1
    const float* s0 = obj + ((size_t)b * NOBJ + p0) * DDIM + k;
    const float* s1 = obj + ((size_t)b * NOBJ + p1) * DDIM + k;
    float4 x0 = ((const float4*)s0)[0], x1 = ((const float4*)s0)[1];
    float4 y0 = ((const float4*)s1)[0], y1 = ((const float4*)s1)[1];
    unsigned short* d = &sA[row * KP + k];
    d[0]=f2bf(x0.x + sc*y0.x); d[1]=f2bf(x0.y + sc*y0.y);
    d[2]=f2bf(x0.z + sc*y0.z); d[3]=f2bf(x0.w + sc*y0.w);
    d[4]=f2bf(x1.x + sc*y1.x); d[5]=f2bf(x1.y + sc*y1.y);
    d[6]=f2bf(x1.z + sc*y1.z); d[7]=f2bf(x1.w + sc*y1.w);
  }
  __syncthreads();

  const int lane = tid & 31;
  const int l16  = lane & 15;
  const int h    = lane >> 4;
  const int nt   = tid >> 5;

  const v8f z = {0.f,0.f,0.f,0.f,0.f,0.f,0.f,0.f};
  v8f accs[4] = {z, z, z, z};

  for (int k = 0; k < KD; k += 32) {
    const int ka = k + 8 * h;
    const int kb = k + 16 + 8 * h;
    Frag16 a;
    a.q[0] = *(const uint4*)(&sA[l16 * KP + ka]);
    a.q[1] = *(const uint4*)(&sA[l16 * KP + kb]);
#pragma unroll
    for (int j = 0; j < 4; ++j) {
      Frag16 bf;
      const unsigned short* w = WT + (size_t)(nt * 64 + j * 16 + l16) * KD;
      bf.q[0] = *(const uint4*)(w + ka);
      bf.q[1] = *(const uint4*)(w + kb);
      accs[j] = __builtin_amdgcn_wmma_f32_16x16x32_bf16(false, a.bf, false, bf.bf,
                                                        (short)0, accs[j], false, false);
    }
  }
#pragma unroll
  for (int j = 0; j < 4; ++j) {
    const int col = nt * 64 + j * 16 + l16;
    const float bv = bias[col];
#pragma unroll
    for (int i = 0; i < 8; ++i) {
      const int row = mt * 16 + 8 * h + i;
      float v = accs[j][i] + bv;
      v = v > 0.f ? v : 0.f;
      v += bf2f(initrel[(size_t)row * DDIM + col]);
      relfeat[(size_t)row * DDIM + col] = f2bf(v);
    }
  }
}

// ---------------------------------------------------------------------------
// Stages 3/4: bf16-A GEMM. A tile is a pure bf16 copy -> stage it into LDS
// with CDNA5 async loads (global_load_async_to_lds_b128, ASYNCcnt-tracked),
// then consume via ds_load_b128. relu + bias epilogue, bf16 or f32 store.
// ---------------------------------------------------------------------------
template <int KD, int ND, bool OUTF32>
__global__ __launch_bounds__(256)
void fc_gemm_kernel(const unsigned short* __restrict__ A,   // (32768 x KD) bf16
                    const unsigned short* __restrict__ WT,  // (ND x KD) bf16
                    const float* __restrict__ bias,
                    unsigned short* __restrict__ outb,
                    float* __restrict__ outf) {
  constexpr int KP = KD + 8;
  constexpr int JT = ND / (8 * 16);   // col-tiles of 16 per wave
  __shared__ unsigned short sA[16 * KP];

  const int tid = threadIdx.x;
  const int mt  = blockIdx.x;

  // ---- async copy of the 16 x KD bf16 A tile into LDS (16B chunks) ----
  for (int c = tid; c < 16 * (KD / 8); c += 256) {
    const int row = c / (KD / 8);
    const int k   = (c % (KD / 8)) * 8;
    const unsigned short* g = A + (size_t)(mt * 16 + row) * KD + k;
    const unsigned int loff = (unsigned int)(uintptr_t)(&sA[row * KP + k]);
    asm volatile("global_load_async_to_lds_b128 %0, %1, off"
                 :: "v"(loff), "v"(g) : "memory");
  }
  asm volatile("s_wait_asynccnt 0x0" ::: "memory");
  __syncthreads();

  const int lane = tid & 31;
  const int l16  = lane & 15;
  const int h    = lane >> 4;
  const int nt   = tid >> 5;

  const v8f z = {0.f,0.f,0.f,0.f,0.f,0.f,0.f,0.f};
  v8f accs[JT];
#pragma unroll
  for (int j = 0; j < JT; ++j) accs[j] = z;

  for (int k = 0; k < KD; k += 32) {
    const int ka = k + 8 * h;
    const int kb = k + 16 + 8 * h;
    Frag16 a;
    a.q[0] = *(const uint4*)(&sA[l16 * KP + ka]);
    a.q[1] = *(const uint4*)(&sA[l16 * KP + kb]);
#pragma unroll
    for (int j = 0; j < JT; ++j) {
      Frag16 bf;
      const unsigned short* w = WT + (size_t)(nt * JT * 16 + j * 16 + l16) * KD;
      bf.q[0] = *(const uint4*)(w + ka);
      bf.q[1] = *(const uint4*)(w + kb);
      accs[j] = __builtin_amdgcn_wmma_f32_16x16x32_bf16(false, a.bf, false, bf.bf,
                                                        (short)0, accs[j], false, false);
    }
  }
#pragma unroll
  for (int j = 0; j < JT; ++j) {
    const int col = nt * JT * 16 + j * 16 + l16;
    const float bv = bias[col];
#pragma unroll
    for (int i = 0; i < 8; ++i) {
      const int row = mt * 16 + 8 * h + i;
      float v = accs[j][i] + bv;
      v = v > 0.f ? v : 0.f;
      if (OUTF32) outf[(size_t)row * ND + col] = v;
      else        outb[(size_t)row * ND + col] = f2bf(v);
    }
  }
}

// ---------------------------------------------------------------------------
// Head: logits = h2 @ fc3_w + fc3_b, softmax over C=7 (too narrow for WMMA).
// ---------------------------------------------------------------------------
__global__ __launch_bounds__(256)
void head_kernel(const float* __restrict__ h2, const float* __restrict__ w,
                 const float* __restrict__ b, float* __restrict__ out) {
  __shared__ float sw[128 * CCLS];
  __shared__ float sb[CCLS];
  for (int i = threadIdx.x; i < 128 * CCLS; i += blockDim.x) sw[i] = w[i];
  if (threadIdx.x < CCLS) sb[threadIdx.x] = b[threadIdx.x];
  __syncthreads();

  const int row = blockIdx.x * blockDim.x + threadIdx.x;
  const float* __restrict__ hr = h2 + (size_t)row * 128;

  float logit[CCLS];
#pragma unroll
  for (int c = 0; c < CCLS; ++c) logit[c] = sb[c];
  for (int k = 0; k < 128; ++k) {
    const float x = hr[k];
#pragma unroll
    for (int c = 0; c < CCLS; ++c) logit[c] = fmaf(x, sw[k * CCLS + c], logit[c]);
  }
  float mx = logit[0];
#pragma unroll
  for (int c = 1; c < CCLS; ++c) mx = fmaxf(mx, logit[c]);
  float sum = 0.f;
#pragma unroll
  for (int c = 0; c < CCLS; ++c) { logit[c] = __expf(logit[c] - mx); sum += logit[c]; }
  const float inv = 1.f / sum;
#pragma unroll
  for (int c = 0; c < CCLS; ++c) out[(size_t)row * CCLS + c] = logit[c] * inv;
}

// ---------------------------------------------------------------------------
extern "C" void kernel_launch(void* const* d_in, const int* in_sizes, int n_in,
                              void* d_out, int out_size, void* d_ws, size_t ws_size,
                              hipStream_t stream) {
  (void)in_sizes; (void)n_in; (void)out_size; (void)ws_size;

  const float* obj    = (const float*)d_in[0];
  const int*   pairs  = (const int*)  d_in[1];
  const float* fuse_w = (const float*)d_in[2];
  const float* fuse_b = (const float*)d_in[3];
  const float* gcn_w  = (const float*)d_in[4];
  const float* gcn_b  = (const float*)d_in[5];
  const float* fc1_w  = (const float*)d_in[6];
  const float* fc1_b  = (const float*)d_in[7];
  const float* fc2_w  = (const float*)d_in[8];
  const float* fc2_b  = (const float*)d_in[9];
  const float* fc3_w  = (const float*)d_in[10];
  const float* fc3_b  = (const float*)d_in[11];
  float* out = (float*)d_out;

  uint8_t* ws = (uint8_t*)d_ws;
  size_t off = 0;
  auto walloc = [&](size_t bytes) -> uint8_t* {
    off = (off + 255) & ~(size_t)255;
    uint8_t* p = ws + off;
    off += bytes;
    return p;
  };
  unsigned short* wt_fuse = (unsigned short*)walloc((size_t)DDIM * 2 * DDIM * 2); // (512x1024)
  unsigned short* wt_gcn  = (unsigned short*)walloc((size_t)DDIM * DDIM * 2);     // (512x512)
  unsigned short* wt_fc1  = (unsigned short*)walloc((size_t)256 * DDIM * 2);      // (256x512)
  unsigned short* wt_fc2  = (unsigned short*)walloc((size_t)128 * 256 * 2);       // (128x256)
  unsigned short* initrel = (unsigned short*)walloc((size_t)MROWS * DDIM * 2);
  unsigned short* relfeat = (unsigned short*)walloc((size_t)MROWS * DDIM * 2);
  unsigned short* h1      = (unsigned short*)walloc((size_t)MROWS * 256 * 2);
  float*          h2      = (float*)         walloc((size_t)MROWS * 128 * 4);

  // Weight prep (transpose + bf16 convert)
  transpose_bf16_kernel<<<(2 * DDIM * DDIM + 255) / 256, 256, 0, stream>>>(fuse_w, wt_fuse, 2 * DDIM, DDIM);
  transpose_bf16_kernel<<<(DDIM * DDIM + 255) / 256, 256, 0, stream>>>(gcn_w, wt_gcn, DDIM, DDIM);
  transpose_bf16_kernel<<<(DDIM * 256 + 255) / 256, 256, 0, stream>>>(fc1_w, wt_fc1, DDIM, 256);
  transpose_bf16_kernel<<<(256 * 128 + 255) / 256, 256, 0, stream>>>(fc2_w, wt_fc2, 256, 128);

  // GEMM pipeline: one block per 16-row M tile, LDS-staged A, 8 waves over N.
  fuse_gemm_kernel<<<MROWS / 16, 256, 0, stream>>>(obj, pairs, wt_fuse, fuse_b, initrel);
  gcn_gemm_kernel<<<MROWS / 16, 256, 0, stream>>>(obj, pairs, wt_gcn, gcn_b, initrel, relfeat);
  fc_gemm_kernel<512, 256, false><<<MROWS / 16, 256, 0, stream>>>(relfeat, wt_fc1, fc1_b, h1, nullptr);
  fc_gemm_kernel<256, 128, true ><<<MROWS / 16, 256, 0, stream>>>(h1, wt_fc2, fc2_b, nullptr, h2);

  // fc3 + softmax
  head_kernel<<<MROWS / 256, 256, 0, stream>>>(h2, fc3_w, fc3_b, out);
}
